// PI0Pytorch_35089882808806
// MI455X (gfx1250) — matrix operand order repaired and checked by reading, not compiled, not run on hardware
//
#include <hip/hip_runtime.h>
#include <hip/hip_bf16.h>

typedef _Float16 v16h  __attribute__((ext_vector_type(16)));
typedef float    v8f   __attribute__((ext_vector_type(8)));
typedef float    f32x4 __attribute__((ext_vector_type(4)));
typedef _Float16 f16x4 __attribute__((ext_vector_type(4)));

#define B_    8
#define H_    16
#define N_    1024
#define D_    128
#define KTILE 32      // keys per inner tile (== WMMA K for the PV matmul)
#define QTILE 256     // q rows per workgroup: 8 waves x 2 stripes x 16 rows
#define NWAVES 8
#define NEG_INF (-1.0e30f)

// D = A*B + C, 16x16x32 f16 -> f32
__device__ __forceinline__ v8f wmma_f16(v16h a, v16h b, v8f c) {
    return __builtin_amdgcn_wmma_f32_16x16x32_f16(
        /*neg_a=*/false, a, /*neg_b=*/false, b,
        /*c_mod=*/(short)0, c, /*reuse_a=*/false, /*reuse_b=*/false);
}

// 16-bit A-matrix (16x32) in-lane half index, per ISA 7.12.2 (two runs of 8).
__device__ __forceinline__ int aIdx(int j, int hi) {
    int v = j >> 1, p = j & 1;
    return (v < 4) ? (2 * v + p + 8 * hi) : (16 + 2 * (v - 4) + p + 8 * hi);
}
// 16-bit B-matrix (32x16): lane = N; same two-runs-of-8 K order as the A-matrix.
__device__ __forceinline__ int bIdx(int j, int hi) { return (j < 8) ? (j + 8 * hi) : (16 + 8 * hi + (j - 8)); }

__device__ __forceinline__ float redMax16(float x) {
    x = fmaxf(x, __shfl_xor(x, 1, 32));
    x = fmaxf(x, __shfl_xor(x, 2, 32));
    x = fmaxf(x, __shfl_xor(x, 4, 32));
    x = fmaxf(x, __shfl_xor(x, 8, 32));
    return x;
}

__global__ __launch_bounds__(256, 1) void
pi0_prefix_attn_kernel(const float* __restrict__ qg,
                       const float* __restrict__ kg,
                       const float* __restrict__ vg,
                       const unsigned char* __restrict__ pad,
                       float* __restrict__ outg,
                       const int* __restrict__ p_auxs,
                       const int* __restrict__ p_auxe,
                       const int* __restrict__ p_pref) {
    // double-buffered K/V tiles + per-wave P scratch: 32 + 16 = 48 KB (of 320 KB)
    __shared__ _Float16 ks[2 * KTILE * D_];
    __shared__ _Float16 vs[2 * KTILE * D_];
    __shared__ _Float16 ps[NWAVES * 32 * KTILE];

    const int tid  = threadIdx.x;
    const int wave = tid >> 5;
    const int lane = tid & 31;
    const int nn   = lane & 15;     // N index (C/B layouts)
    const int hi   = lane >> 4;     // lane-half (row/k split)

    const int qt = blockIdx.x;
    const int hh = blockIdx.y;
    const int bb = blockIdx.z;

    const int auxs = p_auxs[0];
    const int auxe = p_auxe[0];
    const int pref = p_pref[0];

    const size_t plane = ((size_t)bb * H_ + hh) * (size_t)N_ * D_;
    const float* qp = qg + plane;
    const float* kp = kg + plane;
    const float* vp = vg + plane;
    float*       op = outg + plane;
    const unsigned char* padb = pad + bb * N_;

    const int qb0 = qt * QTILE + wave * 32;          // wave's first q row
    // boundaries (768/992/1024) are 16-aligned: one category per 16-row stripe
    int cat[2];
#pragma unroll
    for (int s = 0; s < 2; ++s) {
        int qs = qb0 + s * 16;
        cat[s] = (qs >= auxs) ? 2 : (qs >= pref ? 1 : 0);
    }

    // scale * log2(e), folded into the Q fragments (softmax in exp2 domain)
    const float sl2e = 0.08838834764831845f * 1.4426950408889634f;

    // ---- Q: 2 stripes x 4 A-fragments; vectorized f32x4 loads, scale folded ----
    v16h qa[2][4];
#pragma unroll
    for (int s = 0; s < 2; ++s) {
        const float* qrow = qp + (size_t)(qb0 + s * 16 + nn) * D_;
#pragma unroll
        for (int f = 0; f < 4; ++f) {
            const float* base = qrow + f * 32 + 8 * hi;
            f32x4 r0 = *(const f32x4*)(base);          // k = 8hi+0..3
            f32x4 r1 = *(const f32x4*)(base + 4);      // k = 8hi+4..7
            f32x4 r2 = *(const f32x4*)(base + 16);     // k = 16+8hi+0..3
            f32x4 r3 = *(const f32x4*)(base + 20);     // k = 16+8hi+4..7
            f16x4 h0 = __builtin_convertvector(r0 * sl2e, f16x4);
            f16x4 h1 = __builtin_convertvector(r1 * sl2e, f16x4);
            f16x4 h2 = __builtin_convertvector(r2 * sl2e, f16x4);
            f16x4 h3 = __builtin_convertvector(r3 * sl2e, f16x4);
#pragma unroll
            for (int j = 0; j < 4; ++j) {
                qa[s][f][j]      = h0[j];
                qa[s][f][j + 4]  = h1[j];
                qa[s][f][j + 8]  = h2[j];
                qa[s][f][j + 12] = h3[j];
            }
        }
    }

    v16h ones;
#pragma unroll
    for (int j = 0; j < 16; ++j) ones[j] = (_Float16)1.0f;

    float mrow[2][8];
    v8f o[2][8];
    v8f lacc[2];
#pragma unroll
    for (int s = 0; s < 2; ++s) {
#pragma unroll
        for (int r = 0; r < 8; ++r) { mrow[s][r] = NEG_INF; lacc[s][r] = 0.0f; }
#pragma unroll
        for (int f = 0; f < 8; ++f)
#pragma unroll
            for (int r = 0; r < 8; ++r) o[s][f][r] = 0.0f;
    }

    // ---- workgroup-uniform trip count (needed keys are a contiguous prefix) ----
    int wgMax = 0;
    {
        int qw0 = qt * QTILE, qw1 = qw0 + QTILE - 1;
        if (qw0 < pref) wgMax = pref;
        if (qw1 >= pref && qw0 < auxs && wgMax < auxs) wgMax = auxs;
        if (qw1 >= auxs && wgMax < auxe) wgMax = auxe;
        if (wgMax > N_) wgMax = N_;
    }
    const int NT = (wgMax + KTILE - 1) / KTILE;

    const int psBase = wave * 32 * KTILE;

    // ---- tile body; staging registers passed explicitly (parity-unrolled) ----
    auto body = [&](int t, f32x4 (&sk)[4], f32x4 (&sv)[4],
                    f32x4 (&nk)[4], f32x4 (&nv)[4]) {
        const int kb  = t * KTILE;
        const int buf = (t & 1) * (KTILE * D_);

        // stage tile t (sk/sv registers -> f16 LDS)
#pragma unroll
        for (int i = 0; i < 4; ++i) {
            int e = (tid + 256 * i) * 4;
            *(f16x4*)(ks + buf + e) = __builtin_convertvector(sk[i], f16x4);
            *(f16x4*)(vs + buf + e) = __builtin_convertvector(sv[i], f16x4);
        }
        __syncthreads();

        // issue global loads for tile t+1 into the OTHER register set
        if (t + 1 < NT) {
            const float* ktg = kp + (size_t)(kb + KTILE) * D_;
            const float* vtg = vp + (size_t)(kb + KTILE) * D_;
#pragma unroll
            for (int i = 0; i < 4; ++i) {
                int e = (tid + 256 * i) * 4;
                nk[i] = *(const f32x4*)(ktg + e);
                nv[i] = *(const f32x4*)(vtg + e);
            }
        }
        if (t + 2 < NT) {   // global_prefetch_b8 two tiles ahead
            __builtin_prefetch(kp + (size_t)(kb + 2 * KTILE) * D_ +
                               (tid >> 3) * D_ + (tid & 7) * 16, 0, 1);
            __builtin_prefetch(vp + (size_t)(kb + 2 * KTILE) * D_ +
                               (tid >> 3) * D_ + (tid & 7) * 16, 0, 1);
        }

        // wave-uniform subtile masks per stripe (boundaries are 16-aligned)
        bool s0ok[2], s1ok[2];
#pragma unroll
        for (int s = 0; s < 2; ++s) {
            int k0 = kb, k1 = kb + 16;
            if (cat[s] == 2) {
                s0ok[s] = (k0 < pref) || (k0 >= auxs && k0 < auxe);
                s1ok[s] = (k1 < pref) || (k1 >= auxs && k1 < auxe);
            } else if (cat[s] == 1) {
                s0ok[s] = k0 < auxs; s1ok[s] = k1 < auxs;
            } else {
                s0ok[s] = k0 < pref; s1ok[s] = k1 < pref;
            }
        }
        const bool stripeOk0 = s0ok[0] || s1ok[0];
        const bool stripeOk1 = s0ok[1] || s1ok[1];
        if (!(stripeOk0 || stripeOk1)) return;

        const _Float16* kt = ks + buf;
        const _Float16* vt = vs + buf;

        // ---- S = Q @ K^T : 16 WMMAs; ping-pong bk regs hide LDS latency ----
        auto loadBK = [&](int f, v16h& b0, v16h& b1) {
#pragma unroll
            for (int j = 0; j < 16; ++j) {
                int d = f * 32 + bIdx(j, hi);
                b0[j] = kt[nn * D_ + d];
                b1[j] = kt[(nn + 16) * D_ + d];
            }
        };
        v8f sA[2] = {{0, 0, 0, 0, 0, 0, 0, 0}, {0, 0, 0, 0, 0, 0, 0, 0}};
        v8f sB[2] = {{0, 0, 0, 0, 0, 0, 0, 0}, {0, 0, 0, 0, 0, 0, 0, 0}};
        {
            v16h bk0c, bk1c, bk0n, bk1n;
            loadBK(0, bk0c, bk1c);
#pragma unroll
            for (int f = 0; f < 4; ++f) {
                if (f < 3) loadBK(f + 1, bk0n, bk1n);
                sA[0] = wmma_f16(qa[0][f], bk0c, sA[0]);
                sA[1] = wmma_f16(qa[1][f], bk0c, sA[1]);
                sB[0] = wmma_f16(qa[0][f], bk1c, sB[0]);
                sB[1] = wmma_f16(qa[1][f], bk1c, sB[1]);
                bk0c = bk0n; bk1c = bk1n;   // SSA rename (loop fully unrolled)
            }
        }

        // ---- online softmax per stripe; deferred rescale via wave vote ----
        const bool pk0 = padb[kb + nn] != 0;
        const bool pk1 = padb[kb + 16 + nn] != 0;
#pragma unroll
        for (int s = 0; s < 2; ++s) {
            if (!(s0ok[s] || s1ok[s])) continue;   // wave-uniform
            float x0[8], x1[8], mx[8];
#pragma unroll
            for (int r = 0; r < 8; ++r) {
                x0[r] = (s0ok[s] && pk0) ? sA[s][r] : NEG_INF;
                x1[r] = (s1ok[s] && pk1) ? sB[s][r] : NEG_INF;
                mx[r] = redMax16(fmaxf(x0[r], x1[r]));
            }
            bool upd = false;
#pragma unroll
            for (int r = 0; r < 8; ++r) upd |= (mx[r] > mrow[s][r]);
            if (__any((int)upd)) {    // rare: some row's max advanced
#pragma unroll
                for (int r = 0; r < 8; ++r) {
                    float mn = fmaxf(mrow[s][r], mx[r]);
                    float al = exp2f(mrow[s][r] - mn);
                    mrow[s][r] = mn;
                    lacc[s][r] *= al;
#pragma unroll
                    for (int f = 0; f < 8; ++f) o[s][f][r] *= al;
                }
            }
#pragma unroll
            for (int r = 0; r < 8; ++r) {
                float p0 = exp2f(x0[r] - mrow[s][r]);   // masked -> exp2(-huge)=0
                float p1 = exp2f(x1[r] - mrow[s][r]);
                int mloc = s * 16 + r + 8 * hi;
                ps[psBase + mloc * KTILE + nn]      = (_Float16)p0;
                ps[psBase + mloc * KTILE + 16 + nn] = (_Float16)p1;
            }
        }
        asm volatile("s_wait_dscnt 0" ::: "memory");   // wave-local LDS ordering

        // ---- P -> A layout; O += P @ V; ping-pong bv regs; l += P @ 1 ----
        v16h pa[2];
#pragma unroll
        for (int s = 0; s < 2; ++s)
#pragma unroll
            for (int j = 0; j < 16; ++j)
                pa[s][j] = ps[psBase + (s * 16 + nn) * KTILE + aIdx(j, hi)];
        auto loadBV = [&](int f, v16h& b) {
#pragma unroll
            for (int j = 0; j < 16; ++j)
                b[j] = vt[bIdx(j, hi) * D_ + f * 16 + nn];
        };
        {
            v16h bvc, bvn;
            loadBV(0, bvc);
#pragma unroll
            for (int f = 0; f < 8; ++f) {
                if (f < 7) loadBV(f + 1, bvn);
                if (stripeOk0) o[0][f] = wmma_f16(pa[0], bvc, o[0][f]);
                if (stripeOk1) o[1][f] = wmma_f16(pa[1], bvc, o[1][f]);
                bvc = bvn;                  // SSA rename (loop fully unrolled)
            }
        }
        if (stripeOk0) lacc[0] = wmma_f16(pa[0], ones, lacc[0]);
        if (stripeOk1) lacc[1] = wmma_f16(pa[1], ones, lacc[1]);
    };

    // ---- software-pipelined loop, ping-pong staging register sets ----
    f32x4 kregA[4], vregA[4], kregB[4], vregB[4];
    if (NT > 0) {
#pragma unroll
        for (int i = 0; i < 4; ++i) {
            int e = (tid + 256 * i) * 4;
            kregA[i] = *(const f32x4*)(kp + e);
            vregA[i] = *(const f32x4*)(vp + e);
        }
    }
    for (int t = 0; t < NT; t += 2) {
        body(t, kregA, vregA, kregB, vregB);
        if (t + 1 < NT) body(t + 1, kregB, vregB, kregA, vregA);
    }

    // ---- epilogue: O / l ----
#pragma unroll
    for (int s = 0; s < 2; ++s) {
        float rl[8];
#pragma unroll
        for (int r = 0; r < 8; ++r)
            rl[r] = (lacc[s][r] > 0.0f) ? (1.0f / lacc[s][r]) : 0.0f;
#pragma unroll
        for (int f = 0; f < 8; ++f)
#pragma unroll
            for (int r = 0; r < 8; ++r) {
                int row = qb0 + s * 16 + r + 8 * hi;
                int col = f * 16 + nn;
                op[(size_t)row * D_ + col] = o[s][f][r] * rl[r];
            }
    }
}

extern "C" void kernel_launch(void* const* d_in, const int* in_sizes, int n_in,
                              void* d_out, int out_size, void* d_ws, size_t ws_size,
                              hipStream_t stream) {
    // setup_inputs order:
    // 0:q 1:k 2:v 3:pad_masks 4:att_masks 5:aux_pad_masks
    // 6:aux_start 7:aux_end 8:prefix_len 9:suffix_len
    const float* q = (const float*)d_in[0];
    const float* k = (const float*)d_in[1];
    const float* v = (const float*)d_in[2];
    const unsigned char* pad = (const unsigned char*)d_in[3];
    const int* auxs = (const int*)d_in[6];
    const int* auxe = (const int*)d_in[7];
    const int* pref = (const int*)d_in[8];
    float* out = (float*)d_out;

    dim3 grid(N_ / QTILE, H_, B_);   // 4 x 16 x 8 = 512 workgroups
    dim3 block(256);                 // 8 wave32 waves, 32 q rows each
    pi0_prefix_attn_kernel<<<grid, block, 0, stream>>>(q, k, v, pad, out,
                                                       auxs, auxe, pref);
}
